// CollaborativeWaterfallMoE_74105365725746
// MI455X (gfx1250) — matrix operand order repaired and checked by reading, not compiled, not run on hardware
//
#include <hip/hip_runtime.h>
#include <hip/hip_bf16.h>
#include <cstdint>
#include <cmath>

// ---------------------------------------------------------------------------
// CollaborativeWaterfallMoE forward for MI455X (gfx1250, wave32, WMMA).
// Convs = implicit GEMM on v_wmma_f32_16x16x32_bf16, NHWC bf16 activations.
// v2: 2x2 register blocking per wave (32oc x 32pix, 4 accumulators) so each
// WMMA costs only 1 global b128 + 1 LDS b128 per lane; weights pre-packed to
// [tap][oc][ic] bf16 so the LDS stage is pure coalesced b128 copies.
// ---------------------------------------------------------------------------

typedef __bf16 bf16;
typedef __bf16 v16bf __attribute__((ext_vector_type(16)));
typedef float  v8f   __attribute__((ext_vector_type(8)));

#define BN_RSQ    0.9999950000374997f  /* 1/sqrt(1 + 1e-5) */
#define LOG_EPS  -20.72326583694641f   /* log(1e-9) */
#define T_INV     10.0f                /* 1/T, T = 0.1 */
#define CLIP_V    1e9f
#define CAPACITY  128
#define NB        512                  /* batch */

// ============== weight pack: OIHW f32 -> [tap][oc][ic] bf16 =================
__global__ void pack_weights(const float* __restrict__ w, bf16* __restrict__ wp,
                             int OC, int IC, int total)
{
  int idx = blockIdx.x * 256 + threadIdx.x;
  if (idx >= total) return;
  int ic = idx % IC;
  int t  = idx / IC;
  int oc = t % OC;
  int tap = t / OC;
  wp[idx] = (bf16)w[((size_t)oc * IC + ic) * 9 + tap];
}

// ============================ conv1 (direct) ================================
// x: NCHW f32 [512,3,64,64] -> h1: NHWC bf16 [512,64,64,64], bias+BN+ReLU fused
__global__ void __launch_bounds__(256)
conv1_direct(const float* __restrict__ x, const float* __restrict__ w,
             const float* __restrict__ bias, const float* __restrict__ g,
             const float* __restrict__ be, bf16* __restrict__ out)
{
  int idx = blockIdx.x * 256 + threadIdx.x;           // (pixel, oc) NHWC order
  int oc  = idx & 63;
  int p   = idx >> 6;
  int xx  = p & 63;
  int yy  = (p >> 6) & 63;
  int b   = p >> 12;
  float acc = bias[oc];
  #pragma unroll
  for (int ic = 0; ic < 3; ++ic)
    #pragma unroll
    for (int ky = 0; ky < 3; ++ky) {
      int iy = yy + ky - 1;
      if (iy < 0 || iy > 63) continue;
      #pragma unroll
      for (int kx = 0; kx < 3; ++kx) {
        int ix = xx + kx - 1;
        if (ix < 0 || ix > 63) continue;
        acc = fmaf(x[(((size_t)b * 3 + ic) * 64 + iy) * 64 + ix],
                   w[((oc * 3 + ic) * 3 + ky) * 3 + kx], acc);
      }
    }
  float v = fmaf(acc, g[oc] * BN_RSQ, be[oc]);
  out[(size_t)p * 64 + oc] = (bf16)fmaxf(v, 0.0f);
}

// ====================== WMMA implicit-GEMM conv3x3 ==========================
// in : NHWC bf16 [NP = B*H*W, IC], wpack: [tap][OC][IC] bf16
// out: NHWC bf16 [NP, OC]; epilogue = (conv+bias)*g/sqrt(1+eps)+beta, ReLU.
// Grid: (NP/128, OC/32). Block: 128 (4 waves; each wave: 32oc x 32pix tile).
template <int IC>
__global__ void __launch_bounds__(128)
conv3_wmma(const bf16* __restrict__ in, const bf16* __restrict__ wpack,
           const float* __restrict__ bias, const float* __restrict__ gamma,
           const float* __restrict__ beta, bf16* __restrict__ out,
           int H, int W, int OC)
{
  __shared__ bf16 Alds[9 * 32 * IC];                  // [tap][32 oc][ic]
  const int tid  = threadIdx.x;
  const int lane = tid & 31;
  const int wave = tid >> 5;
  const int oc0  = blockIdx.y * 32;

  // Stage packed weight panel: per tap, rows [oc0, oc0+32) are contiguous.
  constexpr int NCHUNK = 9 * 32 * IC / 8;             // uint4 chunks
  for (int i = tid; i < NCHUNK; i += 128) {
    int e0  = i * 8;
    int tap = e0 / (32 * IC);
    int rem = e0 - tap * 32 * IC;                     // offset within [32][IC]
    ((uint4*)Alds)[i] =
        *(const uint4*)(wpack + ((size_t)tap * OC + oc0) * IC + rem);
  }
  __syncthreads();

  const int n0 = (blockIdx.x * 4 + wave) * 32;        // 32 pixels / wave
  const int nl = lane & 15;                           // N index within tile
  const int hi = (lane >> 4) & 1;                     // lane half

  int ox0[2], oy[2], bb[2];
  #pragma unroll
  for (int ni = 0; ni < 2; ++ni) {                    // each 16-pixel subtile
    int ns  = n0 + ni * 16;                           // lies in a single row
    ox0[ni] = ns % W;
    int t1  = ns / W;
    oy[ni]  = t1 % H;
    bb[ni]  = t1 / H;
  }

  v8f acc[2][2] = {{{}, {}}, {{}, {}}};               // [mi][ni]

  #pragma unroll
  for (int tap = 0; tap < 9; ++tap) {
    const int dy = tap / 3 - 1, dx = tap % 3 - 1;
    bool ok[2];
    const bf16* bp[2];
    bool anyrow = false;
    #pragma unroll
    for (int ni = 0; ni < 2; ++ni) {
      int iy = oy[ni] + dy;
      bool rowok = (iy >= 0) && (iy < H);             // uniform per subtile
      anyrow |= rowok;
      int ix = ox0[ni] + nl + dx;
      ok[ni] = rowok && (ix >= 0) && (ix < W);        // per-lane halo predicate
      bp[ni] = in + ((size_t)(bb[ni] * H + iy) * W + ix) * IC + hi * 16;
    }
    if (!anyrow) continue;                            // uniform per wave

    #pragma unroll
    for (int icc = 0; icc < IC / 32; ++icc) {
      union { uint4 u[2]; v16bf v; } Bm[2];
      #pragma unroll
      for (int ni = 0; ni < 2; ++ni) {
        uint4 b0 = {0, 0, 0, 0}, b1 = {0, 0, 0, 0};
        if (ok[ni]) {
          const uint4* p = (const uint4*)(bp[ni] + icc * 32);
          b0 = p[0];
          b1 = p[1];
        }
        Bm[ni].u[0] = b0; Bm[ni].u[1] = b1;
      }
      union { uint4 u[2]; v16bf v; } Am[2];
      #pragma unroll
      for (int mi = 0; mi < 2; ++mi) {
        const int abase = (tap * 32 + mi * 16 + nl) * IC + icc * 32 + hi * 16;
        Am[mi].u[0] = *(const uint4*)(&Alds[abase]);
        Am[mi].u[1] = *(const uint4*)(&Alds[abase + 8]);
      }
      #pragma unroll
      for (int mi = 0; mi < 2; ++mi)
        #pragma unroll
        for (int ni = 0; ni < 2; ++ni)
          acc[mi][ni] = __builtin_amdgcn_wmma_f32_16x16x32_bf16(
              false, Am[mi].v, false, Bm[ni].v, (short)0, acc[mi][ni],
              false, false);
    }
  }

  // Epilogue: lane holds 8 consecutive oc per (mi,ni) for pixel n0+ni*16+nl.
  const int m0 = hi * 8;
  #pragma unroll
  for (int mi = 0; mi < 2; ++mi) {
    #pragma unroll
    for (int ni = 0; ni < 2; ++ni) {
      size_t pix = (size_t)(n0 + ni * 16) + nl;
      union { uint4 u; bf16 h[8]; } o;
      #pragma unroll
      for (int r = 0; r < 8; ++r) {
        int oc  = oc0 + mi * 16 + m0 + r;
        float v = fmaf(acc[mi][ni][r] + bias[oc], gamma[oc] * BN_RSQ, beta[oc]);
        o.h[r]  = (bf16)fmaxf(v, 0.0f);
      }
      *(uint4*)(out + pix * (size_t)OC + oc0 + mi * 16 + m0) = o.u;
    }
  }
}

// ============================ maxpool 2x2 (NHWC) ============================
__global__ void __launch_bounds__(256)
maxpool_kernel(const bf16* __restrict__ in, bf16* __restrict__ out,
               int C, int H, int W, int total)
{
  int idx = blockIdx.x * 256 + threadIdx.x;
  if (idx >= total) return;
  int c = idx % C;
  int p = idx / C;
  int Wo = W >> 1, Ho = H >> 1;
  int xo = p % Wo; p /= Wo;
  int yo = p % Ho;
  int b  = p / Ho;
  size_t base = ((size_t)(b * H + 2 * yo) * W + 2 * xo) * C + c;
  float m = (float)in[base];
  m = fmaxf(m, (float)in[base + C]);
  m = fmaxf(m, (float)in[base + (size_t)W * C]);
  m = fmaxf(m, (float)in[base + (size_t)W * C + C]);
  out[idx] = (bf16)m;
}

// ==================== spatial mean (NHWC bf16 -> f32) =======================
__global__ void meanpool_kernel(const bf16* __restrict__ in, float* __restrict__ outp,
                                int P, int C)
{
  int c = threadIdx.x;
  int b = blockIdx.x;
  const bf16* base = in + (size_t)b * P * C;
  float s = 0.0f;
  for (int p = 0; p < P; ++p) s += (float)base[p * C + c];
  outp[b * C + c] = s / (float)P;
}

// =================== scorer fold: weff[e,c], beff[e] ========================
// scores = flat @ weff^T + beff  (the two scorer einsums are linear)
__global__ void weff_kernel(const float* __restrict__ spw, const float* __restrict__ spb,
                            const float* __restrict__ shw, const float* __restrict__ shb,
                            float* __restrict__ weff, float* __restrict__ beff)
{
  int tid = threadIdx.x;                               // 512 threads, 1 block
  int e = tid >> 7, c = tid & 127;
  float a = 0.0f;
  for (int h = 0; h < 128; ++h)
    a = fmaf(spw[(e * 128 + h) * 128 + c], shw[e * 128 + h], a);
  weff[tid] = a;
  if (tid < 4) {
    float bsum = shb[tid];
    for (int h = 0; h < 128; ++h)
      bsum = fmaf(spb[tid * 128 + h], shw[tid * 128 + h], bsum);
    beff[tid] = bsum;
  }
}

__global__ void scores_kernel(const float* __restrict__ flat, const float* __restrict__ weff,
                              const float* __restrict__ beff, float* __restrict__ scores)
{
  int idx = blockIdx.x * 256 + threadIdx.x;            // 2048 = 512*4
  int b = idx >> 2, e = idx & 3;
  const float* f = flat + b * 128;
  const float* w = weff + e * 128;
  float a = beff[e];
  for (int c = 0; c < 128; ++c) a = fmaf(f[c], w[c], a);
  scores[idx] = a;
}

__global__ void colmean_kernel(const float* __restrict__ scores, float* __restrict__ smean)
{
  int e = threadIdx.x;
  if (e < 4) {
    float s = 0.0f;
    for (int b = 0; b < NB; ++b) s += scores[b * 4 + e];
    smean[e] = s * (1.0f / NB);
  }
}

// ================= waterfall top-1 routing (1 wave, ballot) =================
__global__ void route_kernel(const float* __restrict__ scores, const float* __restrict__ smean,
                             int* __restrict__ assignExp)
{
  const int lane = threadIdx.x;                        // 32 threads = 1 wave
  float sn[16][4];
  for (int ch = 0; ch < 16; ++ch) {
    int t = ch * 32 + lane;
    #pragma unroll
    for (int e = 0; e < 4; ++e) {
      float v = (scores[t * 4 + e] - smean[e] + LOG_EPS) * T_INV;
      sn[ch][e] = fminf(fmaxf(v, -CLIP_V), CLIP_V);
    }
  }
  int cap[4] = {0, 0, 0, 0};
  unsigned assignedMask = 0;
  for (int it = 0; it < 15; ++it) {
    int quota = 1 << it;
    int bests[16];
    for (int ch = 0; ch < 16; ++ch) {                  // argmax (first max wins)
      int best = 0;
      float bv = (cap[0] >= CAPACITY) ? -INFINITY : sn[ch][0];
      #pragma unroll
      for (int e = 1; e < 4; ++e) {
        float v = (cap[e] >= CAPACITY) ? -INFINITY : sn[ch][e];
        if (v > bv) { bv = v; best = e; }
      }
      bests[ch] = best;
    }
    for (int e = 0; e < 4; ++e) {
      int space = CAPACITY - cap[e];
      if (space > quota) space = quota;
      int cnt = 0;
      for (int ch = 0; ch < 16; ++ch) {                // token-index order scan
        bool want = !((assignedMask >> ch) & 1u) && (bests[ch] == e);
        unsigned m = (unsigned)__ballot(want);
        int rank = cnt + __popc(m & ((1u << lane) - 1u));
        if (want && rank < space) {
          assignedMask |= (1u << ch);
          assignExp[ch * 32 + lane] = e;
        }
        int total = __popc(m);
        int rem = space - cnt; if (rem < 0) rem = 0;
        cnt += (total < rem) ? total : rem;
      }
      cap[e] += cnt;
    }
  }
  int least = 0, cv = cap[0];
  #pragma unroll
  for (int e = 1; e < 4; ++e) if (cap[e] < cv) { cv = cap[e]; least = e; }
  for (int ch = 0; ch < 16; ++ch)
    if (!((assignedMask >> ch) & 1u)) assignExp[ch * 32 + lane] = least;
}

// ========== per-token head: z = relu(zf@pw^T+pb); out = z@clw^T+clb =========
__global__ void __launch_bounds__(256)
head_kernel(const float* __restrict__ zf, const int* __restrict__ assignExp,
            const float* __restrict__ pw, const float* __restrict__ pb,
            const float* __restrict__ clw, const float* __restrict__ clb,
            float* __restrict__ out)
{
  __shared__ float z[256];
  int b = blockIdx.x;
  int e = assignExp[b];
  int j = threadIdx.x;
  const float* zrow = zf + ((size_t)e * NB + b) * 256;
  const float* wrow = pw + ((size_t)e * 256 + j) * 256;
  float a = pb[e * 256 + j];
  for (int c = 0; c < 256; ++c) a = fmaf(zrow[c], wrow[c], a);
  z[j] = fmaxf(a, 0.0f);
  __syncthreads();
  if (j < 10) {
    const float* cr = clw + ((size_t)e * 10 + j) * 256;
    float o = clb[e * 10 + j];
    for (int k = 0; k < 256; ++k) o = fmaf(z[k], cr[k], o);
    out[b * 10 + j] = o;
  }
}

// ============================== launcher ====================================
extern "C" void kernel_launch(void* const* d_in, const int* in_sizes, int n_in,
                              void* d_out, int out_size, void* d_ws, size_t ws_size,
                              hipStream_t stream)
{
  (void)in_sizes; (void)n_in; (void)out_size; (void)ws_size;
  // setup_inputs() flat order: x, then params in dict order.
  const float* x   = (const float*)d_in[0];
  const float* w1  = (const float*)d_in[1];
  const float* b1  = (const float*)d_in[2];
  const float* g1  = (const float*)d_in[3];
  const float* be1 = (const float*)d_in[4];
  const float* w2  = (const float*)d_in[5];
  const float* b2  = (const float*)d_in[6];
  const float* g2  = (const float*)d_in[7];
  const float* be2 = (const float*)d_in[8];
  const float* w3  = (const float*)d_in[9];
  const float* b3  = (const float*)d_in[10];
  const float* g3  = (const float*)d_in[11];
  const float* be3 = (const float*)d_in[12];
  const float* w4  = (const float*)d_in[13];
  const float* b4  = (const float*)d_in[14];
  const float* g4  = (const float*)d_in[15];
  const float* be4 = (const float*)d_in[16];
  const float* cw  = (const float*)d_in[17];
  const float* cb  = (const float*)d_in[18];
  const float* cg  = (const float*)d_in[19];
  const float* cbe = (const float*)d_in[20];
  const float* pw  = (const float*)d_in[21];
  const float* pb  = (const float*)d_in[22];
  const float* clw = (const float*)d_in[23];
  const float* clb = (const float*)d_in[24];
  const float* spw = (const float*)d_in[25];
  const float* spb = (const float*)d_in[26];
  const float* shw = (const float*)d_in[27];
  const float* shb = (const float*)d_in[28];
  // d_in[29] = chw, d_in[30] = chb : unused by the reference forward pass.

  char* ws = (char*)d_ws;
  // Workspace layout (regions reused across phases):
  constexpr size_t SZ_AB    = 268435456;               // 512*64*64*64 bf16
  constexpr size_t OFF_H1   = 0;                       // h1 [512,64,64,64]; later h3
  constexpr size_t OFF_H2   = SZ_AB;                   // h2; later h4
  constexpr size_t OFF_P1   = 2 * SZ_AB;               // p1 [512,32,32,64]; later expert conv [512,16,16,256]
  constexpr size_t OFF_FEAT = OFF_P1 + 67108864;       // features [512,16,16,128] bf16
  constexpr size_t OFF_FLAT = OFF_FEAT + 33554432;     // flat f32 [512,128]
  constexpr size_t OFF_ZF   = OFF_FLAT + 262144;       // zf f32 [4,512,256]
  constexpr size_t OFF_WEFF = OFF_ZF + 2097152;
  constexpr size_t OFF_BEFF = OFF_WEFF + 2048;
  constexpr size_t OFF_SC   = OFF_BEFF + 64;           // scores f32 [512,4]
  constexpr size_t OFF_SM   = OFF_SC + 8192;           // smean f32 [4]
  constexpr size_t OFF_ASG  = OFF_SM + 64;             // assignExp int [512]
  constexpr size_t OFF_WP2  = OFF_ASG + 2048;          // packed w2 bf16 [9,64,64]
  constexpr size_t OFF_WP3  = OFF_WP2 + 73728;         // packed w3 bf16 [9,128,64]
  constexpr size_t OFF_WP4  = OFF_WP3 + 147456;        // packed w4 bf16 [9,128,128]
  constexpr size_t OFF_WPC  = OFF_WP4 + 294912;        // packed cw bf16 [4][9,256,128]

  bf16*  h1   = (bf16*)(ws + OFF_H1);
  bf16*  h2   = (bf16*)(ws + OFF_H2);
  bf16*  p1   = (bf16*)(ws + OFF_P1);
  bf16*  h3   = (bf16*)(ws + OFF_H1);
  bf16*  h4   = (bf16*)(ws + OFF_H2);
  bf16*  feat = (bf16*)(ws + OFF_FEAT);
  bf16*  ec   = (bf16*)(ws + OFF_P1);
  float* flat = (float*)(ws + OFF_FLAT);
  float* zf   = (float*)(ws + OFF_ZF);
  float* weff = (float*)(ws + OFF_WEFF);
  float* beff = (float*)(ws + OFF_BEFF);
  float* sc   = (float*)(ws + OFF_SC);
  float* sm   = (float*)(ws + OFF_SM);
  int*   asg  = (int*)(ws + OFF_ASG);
  bf16*  wp2  = (bf16*)(ws + OFF_WP2);
  bf16*  wp3  = (bf16*)(ws + OFF_WP3);
  bf16*  wp4  = (bf16*)(ws + OFF_WP4);
  bf16*  wpc  = (bf16*)(ws + OFF_WPC);
  float* outp = (float*)d_out;

  // ---- pack conv weights to [tap][oc][ic] bf16 ----
  pack_weights<<<144, 256, 0, stream>>>(w2, wp2, 64, 64, 36864);
  pack_weights<<<288, 256, 0, stream>>>(w3, wp3, 128, 64, 73728);
  pack_weights<<<576, 256, 0, stream>>>(w4, wp4, 128, 128, 147456);
  for (int e = 0; e < 4; ++e)
    pack_weights<<<1152, 256, 0, stream>>>(cw + (size_t)e * 294912,
                                           wpc + (size_t)e * 294912,
                                           256, 128, 294912);

  // ---- shared encoder ----
  conv1_direct<<<524288, 256, 0, stream>>>(x, w1, b1, g1, be1, h1);
  conv3_wmma<64><<<dim3(16384, 2), 128, 0, stream>>>(h1, wp2, b2, g2, be2, h2, 64, 64, 64);
  maxpool_kernel<<<131072, 256, 0, stream>>>(h2, p1, 64, 64, 64, 33554432);
  conv3_wmma<64><<<dim3(4096, 4), 128, 0, stream>>>(p1, wp3, b3, g3, be3, h3, 32, 32, 128);
  conv3_wmma<128><<<dim3(4096, 4), 128, 0, stream>>>(h3, wp4, b4, g4, be4, h4, 32, 32, 128);
  maxpool_kernel<<<65536, 256, 0, stream>>>(h4, feat, 128, 32, 32, 16777216);
  meanpool_kernel<<<NB, 128, 0, stream>>>(feat, flat, 256, 128);

  // ---- scorer (folded linear) + routing ----
  weff_kernel<<<1, 512, 0, stream>>>(spw, spb, shw, shb, weff, beff);
  scores_kernel<<<8, 256, 0, stream>>>(flat, weff, beff, sc);
  colmean_kernel<<<1, 32, 0, stream>>>(sc, sm);
  route_kernel<<<1, 32, 0, stream>>>(sc, sm, asg);

  // ---- per-expert conv heads (dense over batch) + pooling ----
  for (int e = 0; e < 4; ++e) {
    conv3_wmma<128><<<dim3(1024, 8), 128, 0, stream>>>(
        feat, wpc + (size_t)e * 294912, cb + e * 256, cg + e * 256,
        cbe + e * 256, ec, 16, 16, 256);
    meanpool_kernel<<<NB, 256, 0, stream>>>(ec, zf + (size_t)e * NB * 256, 256, 256);
  }

  // ---- projection + classifier for the assigned expert, one-hot combine ----
  head_kernel<<<NB, 256, 0, stream>>>(zf, asg, pw, pb, clw, clb, outp);
}
